// FustionLayer_17179869184529
// MI455X (gfx1250) — matrix operand order repaired and checked by reading, not compiled, not run on hardware
//
#include <hip/hip_runtime.h>
#include <stdint.h>

// Problem dims (fixed by reference)
#define BB 64
#define TT 512
#define VV 256
#define HH 512

// Tiling
#define BM 128
#define BN 128
#define BK 32
#define LDT 40   // LDS row stride in bf16 elements (80B: 16B-aligned, conflict-free)

typedef __attribute__((ext_vector_type(16))) __bf16 v16bf;
typedef __attribute__((ext_vector_type(8)))  __bf16 v8bf;
typedef __attribute__((ext_vector_type(4)))  __bf16 v4bf;
typedef __attribute__((ext_vector_type(8)))  float  v8f;
typedef __attribute__((ext_vector_type(4)))  int    v4i;

union Frag { v16bf v; v8bf h[2]; };

// ---- CDNA5 async global->LDS copy (16B per lane), tracked by ASYNCcnt ----
__device__ __forceinline__ void async_copy16(const void* g, void* l) {
    __builtin_amdgcn_global_load_async_to_lds_b128(
        (__attribute__((address_space(1))) v4i*)(v4i*)g,
        (__attribute__((address_space(3))) v4i*)(v4i*)l,
        0, 0);
}
__device__ __forceinline__ void wait_async0() {
    __builtin_amdgcn_s_wait_asynccnt(0);
}

// ---------------- kernel 0: init flag + convert W (f32) -> bf16 ----------------
__global__ void fusion_init_kernel(const float* __restrict__ W,
                                   __bf16* __restrict__ Wb,
                                   int* __restrict__ flag) {
    int i = blockIdx.x * blockDim.x + threadIdx.x;
    if (i == 0) *flag = 1;
    int base = i * 4;
    if (base < HH * HH) {
        float4 w = *(const float4*)(W + base);
        v4bf o;
        o[0] = (__bf16)w.x; o[1] = (__bf16)w.y; o[2] = (__bf16)w.z; o[3] = (__bf16)w.w;
        *(v4bf*)(Wb + base) = o;
    }
}

// ---------------- kernel 1: P = relu(X @ W^T + b) in bf16 ----------------
// X: [M, H] f32 row-major.  Wb: [H, H] bf16, row d contiguous over h.
__global__ __launch_bounds__(256, 2)
void proj_gemm_kernel(const float*  __restrict__ X,
                      const __bf16* __restrict__ Wb,
                      const float*  __restrict__ bias,
                      __bf16*       __restrict__ P) {
    __shared__ __bf16 ldsA[2][BM * LDT];
    __shared__ __bf16 ldsB[2][BN * LDT];

    const int tid  = threadIdx.x;
    const int lane = tid & 31;
    const int wave = tid >> 5;
    const int waveRow = wave >> 2;   // 0..1 -> 64 rows each
    const int waveCol = wave & 3;    // 0..3 -> 32 cols each
    const int rowBase = blockIdx.y * BM;
    const int colBase = blockIdx.x * BN;
    const int lh = lane & 15, hh = lane >> 4;

    v8f acc[4][2] = {};
    float4 ra[4];

    // A: f32 source, VGPR-staged (needs f32->bf16 convert). 1024 float4 chunks.
    auto loadA = [&](int kb) {
        #pragma unroll
        for (int it = 0; it < 4; ++it) {
            int c = tid + 256 * it;
            int r = c >> 3, cc = (c & 7) * 4;
            ra[it] = *(const float4*)(X + (size_t)(rowBase + r) * HH + kb * BK + cc);
        }
    };
    auto storeA = [&](int buf) {
        #pragma unroll
        for (int it = 0; it < 4; ++it) {
            int c = tid + 256 * it;
            int r = c >> 3, cc = (c & 7) * 4;
            v4bf o;
            o[0] = (__bf16)ra[it].x; o[1] = (__bf16)ra[it].y;
            o[2] = (__bf16)ra[it].z; o[3] = (__bf16)ra[it].w;
            *(v4bf*)(&ldsA[buf][r * LDT + cc]) = o;
        }
    };
    // B: bf16 source, async global->LDS, no VGPR staging. 512 16B chunks.
    auto asyncB = [&](int kb, int buf) {
        #pragma unroll
        for (int it = 0; it < 2; ++it) {
            int c = tid + 256 * it;
            int r = c >> 2, cc = (c & 3) * 8;
            async_copy16(Wb + (size_t)(colBase + r) * HH + kb * BK + cc,
                         &ldsB[buf][r * LDT + cc]);
        }
    };
    auto compute = [&](int buf) {
        Frag af[4], bfg[2];
        #pragma unroll
        for (int mt = 0; mt < 4; ++mt) {
            const __bf16* p = &ldsA[buf][(waveRow * 64 + mt * 16 + lh) * LDT + hh * 8];
            af[mt].h[0] = *(const v8bf*)p;
            af[mt].h[1] = *(const v8bf*)(p + 16);
        }
        #pragma unroll
        for (int nt = 0; nt < 2; ++nt) {
            const __bf16* p = &ldsB[buf][(waveCol * 32 + nt * 16 + lh) * LDT + hh * 8];
            bfg[nt].h[0] = *(const v8bf*)p;
            bfg[nt].h[1] = *(const v8bf*)(p + 16);
        }
        #pragma unroll
        for (int mt = 0; mt < 4; ++mt)
            #pragma unroll
            for (int nt = 0; nt < 2; ++nt)
                acc[mt][nt] = __builtin_amdgcn_wmma_f32_16x16x32_bf16(
                    false, af[mt].v, false, bfg[nt].v, (short)0, acc[mt][nt], false, false);
    };

    const int NK = HH / BK;  // 16
    asyncB(0, 0);
    loadA(0);
    storeA(0);
    wait_async0();
    __syncthreads();
    int buf = 0;
    for (int kb = 0; kb < NK; ++kb) {
        if (kb + 1 < NK) { asyncB(kb + 1, buf ^ 1); loadA(kb + 1); }
        compute(buf);
        if (kb + 1 < NK) storeA(buf ^ 1);
        wait_async0();
        __syncthreads();
        buf ^= 1;
    }

    // epilogue: bias + relu + bf16 store
    #pragma unroll
    for (int mt = 0; mt < 4; ++mt) {
        #pragma unroll
        for (int nt = 0; nt < 2; ++nt) {
            int col = colBase + waveCol * 32 + nt * 16 + lh;
            float bv = bias[col];
            int rbase = rowBase + waveRow * 64 + mt * 16 + hh * 8;
            #pragma unroll
            for (int r = 0; r < 8; ++r) {
                float v = acc[mt][nt][r] + bv;
                v = v > 0.0f ? v : 0.0f;
                P[(size_t)(rbase + r) * HH + col] = (__bf16)v;
            }
        }
    }
}

// ---------------- kernel 2: per-batch logits = Xp_b @ Yp_b^T ----------------
// emits sign bytes + global all-positive flag; fully async tile stage-in.
__global__ __launch_bounds__(256, 2)
void sim_gemm_kernel(const __bf16* __restrict__ Xp,     // [B*T, H]
                     const __bf16* __restrict__ Yp,     // [B*V, H]
                     unsigned char* __restrict__ cross, // [B*T*V]
                     int* __restrict__ flag) {
    __shared__ __bf16 ldsA[2][BM * LDT];
    __shared__ __bf16 ldsB[2][BN * LDT];
    __shared__ int s_ok;

    const int tid  = threadIdx.x;
    const int lane = tid & 31;
    const int wave = tid >> 5;
    const int waveRow = wave >> 2;
    const int waveCol = wave & 3;
    const int b = blockIdx.z;
    const int rowBase = blockIdx.y * BM;   // t
    const int colBase = blockIdx.x * BN;   // v
    const int lh = lane & 15, hh = lane >> 4;

    const __bf16* A  = Xp + (size_t)b * TT * HH;
    const __bf16* Bm = Yp + (size_t)b * VV * HH;

    if (tid == 0) s_ok = 1;

    v8f acc[4][2] = {};

    auto asyncTiles = [&](int kb, int buf) {
        #pragma unroll
        for (int it = 0; it < 2; ++it) {
            int c = tid + 256 * it;
            int r = c >> 2, cc = (c & 3) * 8;
            async_copy16(A  + (size_t)(rowBase + r) * HH + kb * BK + cc,
                         &ldsA[buf][r * LDT + cc]);
            async_copy16(Bm + (size_t)(colBase + r) * HH + kb * BK + cc,
                         &ldsB[buf][r * LDT + cc]);
        }
    };
    auto compute = [&](int buf) {
        Frag af[4], bfg[2];
        #pragma unroll
        for (int mt = 0; mt < 4; ++mt) {
            const __bf16* p = &ldsA[buf][(waveRow * 64 + mt * 16 + lh) * LDT + hh * 8];
            af[mt].h[0] = *(const v8bf*)p;
            af[mt].h[1] = *(const v8bf*)(p + 16);
        }
        #pragma unroll
        for (int nt = 0; nt < 2; ++nt) {
            const __bf16* p = &ldsB[buf][(waveCol * 32 + nt * 16 + lh) * LDT + hh * 8];
            bfg[nt].h[0] = *(const v8bf*)p;
            bfg[nt].h[1] = *(const v8bf*)(p + 16);
        }
        #pragma unroll
        for (int mt = 0; mt < 4; ++mt)
            #pragma unroll
            for (int nt = 0; nt < 2; ++nt)
                acc[mt][nt] = __builtin_amdgcn_wmma_f32_16x16x32_bf16(
                    false, af[mt].v, false, bfg[nt].v, (short)0, acc[mt][nt], false, false);
    };

    const int NK = HH / BK;
    asyncTiles(0, 0);
    wait_async0();
    __syncthreads();
    int buf = 0;
    for (int kb = 0; kb < NK; ++kb) {
        if (kb + 1 < NK) asyncTiles(kb + 1, buf ^ 1);
        compute(buf);
        wait_async0();
        __syncthreads();
        buf ^= 1;
    }

    // epilogue: sigmoid(x)>0.5 <=> x>0 ; min(temp)>0.5 <=> all logits>0
    int ok = 1;
    #pragma unroll
    for (int mt = 0; mt < 4; ++mt) {
        #pragma unroll
        for (int nt = 0; nt < 2; ++nt) {
            int v = colBase + waveCol * 32 + nt * 16 + lh;
            int tbase = rowBase + waveRow * 64 + mt * 16 + hh * 8;
            #pragma unroll
            for (int r = 0; r < 8; ++r) {
                int c = acc[mt][nt][r] > 0.0f;
                ok &= c;
                cross[(size_t)b * TT * VV + (size_t)(tbase + r) * VV + v] = (unsigned char)c;
            }
        }
    }
    if (!ok) atomicAnd(&s_ok, 0);
    __syncthreads();
    if (tid == 0 && s_ok == 0) atomicAnd(flag, 0);
}

// ---------------- kernel 3: assemble [B, 768, 768] block adjacency ----------------
__global__ __launch_bounds__(256)
void assemble_kernel(const int* __restrict__ tadj,   // [B,T,T]
                     const int* __restrict__ vadj,   // [B,V,V]
                     const int* __restrict__ tmask,  // [B,T]
                     const int* __restrict__ vmask,  // [B,V]
                     const unsigned char* __restrict__ cross, // [B,T,V]
                     const int* __restrict__ flag,
                     float* __restrict__ out) {      // [B,768,768]
    const int NT = TT + VV;          // 768
    const int C4 = NT / 4;           // 192
    int idx = blockIdx.x * blockDim.x + threadIdx.x;   // one float4 per thread
    int c4  = idx % C4;
    int row = (idx / C4) % NT;
    int b   = idx / (C4 * NT);
    int col0 = c4 * 4;

    float4 r;
    if (row < TT) {
        if (col0 < TT) {
            const int* p = tadj + ((size_t)b * TT + row) * TT + col0;
            r.x = (p[0] != 0) ? 1.0f : 0.0f;
            r.y = (p[1] != 0) ? 1.0f : 0.0f;
            r.z = (p[2] != 0) ? 1.0f : 0.0f;
            r.w = (p[3] != 0) ? 1.0f : 0.0f;
        } else {
            int v0 = col0 - TT;
            int tm = tmask[b * TT + row];
            int fv = *flag;
            const unsigned char* cp = cross + ((size_t)b * TT + row) * VV + v0;
            float vals[4];
            #pragma unroll
            for (int i = 0; i < 4; ++i) {
                int valid = (tm != 0) && (vmask[b * VV + v0 + i] != 0);
                vals[i] = (valid ? (int)cp[i] : fv) ? 1.0f : 0.0f;
            }
            r.x = vals[0]; r.y = vals[1]; r.z = vals[2]; r.w = vals[3];
        }
    } else {
        int ir = row - TT;
        if (col0 < TT) {
            r.x = r.y = r.z = r.w = 0.0f;
        } else {
            const int* p = vadj + ((size_t)b * VV + ir) * VV + (col0 - TT);
            r.x = (p[0] != 0) ? 1.0f : 0.0f;
            r.y = (p[1] != 0) ? 1.0f : 0.0f;
            r.z = (p[2] != 0) ? 1.0f : 0.0f;
            r.w = (p[3] != 0) ? 1.0f : 0.0f;
        }
    }
    *(float4*)(out + ((size_t)b * NT + row) * NT + col0) = r;
}

// ---------------- launch ----------------
extern "C" void kernel_launch(void* const* d_in, const int* in_sizes, int n_in,
                              void* d_out, int out_size, void* d_ws, size_t ws_size,
                              hipStream_t stream) {
    const float* Xtext = (const float*)d_in[0];
    const float* Ximg  = (const float*)d_in[1];
    const float* W     = (const float*)d_in[2];
    const float* bias  = (const float*)d_in[3];
    const int* tmask   = (const int*)d_in[4];
    const int* vmask   = (const int*)d_in[5];
    const int* tadj    = (const int*)d_in[6];
    const int* vadj    = (const int*)d_in[7];
    float* out = (float*)d_out;

    char* ws = (char*)d_ws;
    int*    flag = (int*)ws;
    __bf16* Wb   = (__bf16*)(ws + 256);
    __bf16* Xp   = (__bf16*)(ws + 256 + (size_t)HH * HH * 2);
    __bf16* Yp   = Xp + (size_t)BB * TT * HH;
    unsigned char* cross = (unsigned char*)(Yp + (size_t)BB * VV * HH);
    // ws usage: 256 + 0.5MB + 32MB + 16MB + 8MB ~= 57MB

    fusion_init_kernel<<<(HH * HH / 4 + 255) / 256, 256, 0, stream>>>(W, Wb, flag);

    dim3 g1(HH / BN, (BB * TT) / BM);   // (4, 256)
    proj_gemm_kernel<<<g1, 256, 0, stream>>>(Xtext, Wb, bias, Xp);

    dim3 g2(HH / BN, (BB * VV) / BM);   // (4, 128)
    proj_gemm_kernel<<<g2, 256, 0, stream>>>(Ximg, Wb, bias, Yp);

    dim3 g3(VV / BN, TT / BM, BB);      // (2, 4, 64)
    sim_gemm_kernel<<<g3, 256, 0, stream>>>(Xp, Yp, cross, flag);

    int total4 = BB * (TT + VV) * ((TT + VV) / 4);   // 9,437,184
    assemble_kernel<<<total4 / 256, 256, 0, stream>>>(tadj, vadj, tmask, vmask, cross, flag, out);
}
